// QuanvolutionNet_65481071396250
// MI455X (gfx1250) — matrix operand order repaired and checked by reading, not compiled, not run on hardware
//
#include <hip/hip_runtime.h>
#include <hip/hip_bf16.h>

// Fused quanvolution net for gfx1250 (MI455X).
// One kernel: quantum patch features (factorized 2x2-qubit real simulation)
// + classical pixel permute -> 16x1568 f16 tile in LDS -> WMMA f16 GEMM vs
// W^T (N padded to 16) -> bias + log_softmax -> (B,10) f32 out.

typedef _Float16 v16h __attribute__((ext_vector_type(16)));
typedef _Float16 v8h  __attribute__((ext_vector_type(8)));
typedef float    v8f  __attribute__((ext_vector_type(8)));

#define KDIM 1568      // feature width = 196*4 quantum + 784 classical
#define NROW 16        // batch rows per workgroup (one WMMA M-tile)
#define NLAYER 4

// RY on qubit0 of a real 2-qubit state s[q0][q1]; c=cos(t/2), s=sin(t/2)
__device__ __forceinline__ void ry_q0(float c, float s,
    float& s00, float& s01, float& s10, float& s11) {
  float n00 = c*s00 - s*s10, n01 = c*s01 - s*s11;
  float n10 = s*s00 + c*s10, n11 = s*s01 + c*s11;
  s00 = n00; s01 = n01; s10 = n10; s11 = n11;
}
// RY on qubit1
__device__ __forceinline__ void ry_q1(float c, float s,
    float& s00, float& s01, float& s10, float& s11) {
  float n00 = c*s00 - s*s01, n01 = s*s00 + c*s01;
  float n10 = c*s10 - s*s11, n11 = s*s10 + c*s11;
  s00 = n00; s01 = n01; s10 = n10; s11 = n11;
}

// Independent 2-qubit subsystem: RY(t0)q0, RY(t1)q1, then per layer
// RY(p[l,wa])q0, RY(p[l,wb])q1, CNOT(q0->q1). Returns <Z0>, <Z1>.
__device__ __forceinline__ void sim2q(float t0, float t1,
    const float* __restrict__ pc, const float* __restrict__ ps,
    int wa, int wb, float& z0, float& z1) {
  float s00 = 1.f, s01 = 0.f, s10 = 0.f, s11 = 0.f;
  float c, s;
  __sincosf(t0 * 0.5f, &s, &c); ry_q0(c, s, s00, s01, s10, s11);
  __sincosf(t1 * 0.5f, &s, &c); ry_q1(c, s, s00, s01, s10, s11);
  #pragma unroll
  for (int l = 0; l < NLAYER; ++l) {
    ry_q0(pc[l*4+wa], ps[l*4+wa], s00, s01, s10, s11);
    ry_q1(pc[l*4+wb], ps[l*4+wb], s00, s01, s10, s11);
    float t = s10; s10 = s11; s11 = t;   // CNOT(q0,q1)
  }
  float a0 = s00*s00, a1 = s01*s01, a2 = s10*s10, a3 = s11*s11;
  z0 = (a0 + a1) - (a2 + a3);
  z1 = (a0 + a2) - (a1 + a3);
}

__global__ __launch_bounds__(256)
void quanv_fused_kernel(const float* __restrict__ x,       // (B, 784)
                        const float* __restrict__ params,  // (4, 4)
                        const float* __restrict__ W,       // (10, 1568)
                        const float* __restrict__ bias,    // (10,)
                        float* __restrict__ out)           // (B, 10)
{
  __shared__ _Float16 sF[NROW * KDIM];   // 50176 B: 16-row feature tile
  __shared__ float sPc[16], sPs[16];     // cos/sin of half param angles
  __shared__ float sAcc[8][NROW][16];    // 8 partial WMMA tiles
  __shared__ float sLog[NROW][16];       // reduced logits

  const int tid = threadIdx.x;
  const int b0  = blockIdx.x * NROW;

  if (tid < 16) {
    float c, s;
    __sincosf(params[tid] * 0.5f, &s, &c);
    sPc[tid] = c; sPs[tid] = s;
  }
  __syncthreads();

  // ---- Phase 1: quantum features (196 patches x 16 rows) ----
  for (int idx = tid; idx < NROW * 196; idx += 256) {
    int row = idx / 196, p = idx % 196;
    int pi = p / 14, pj = p % 14;
    const float* xb = x + (size_t)(b0 + row) * 784;
    float a00 = xb[(2*pi    )*28 + 2*pj    ];
    float a01 = xb[(2*pi    )*28 + 2*pj + 1];
    float a10 = xb[(2*pi + 1)*28 + 2*pj    ];
    float a11 = xb[(2*pi + 1)*28 + 2*pj + 1];
    float z0, z1, z2, z3;
    sim2q(a00, a01, sPc, sPs, 0, 1, z0, z1);   // wires 0,1
    sim2q(a10, a11, sPc, sPs, 2, 3, z2, z3);   // wires 2,3
    _Float16* f = sF + row * KDIM + p * 4;
    f[0] = (_Float16)z0; f[1] = (_Float16)z1;
    f[2] = (_Float16)z2; f[3] = (_Float16)z3;
  }

  // ---- Phase 2: classical permuted features (784 per row) ----
  for (int idx = tid; idx < NROW * 784; idx += 256) {
    int row = idx / 784, f = idx % 784;
    int j  = f % 7;
    int i  = (f / 7) % 7;
    int cc = (f / 49) % 4;
    int aa = f / 196;
    float v = x[(size_t)(b0 + row) * 784 + (4*i + aa) * 28 + (4*j + cc)];
    sF[row * KDIM + 784 + f] = (_Float16)v;
  }
  __syncthreads();

  // ---- Phase 3: WMMA GEMM (16 x 1568) @ (1568 x 16), K split over 8 waves ----
  const int wave = tid >> 5;
  const int lane = tid & 31;
  const int half = lane >> 4;     // 0 = lanes 0-15, 1 = lanes 16-31
  const int l15  = lane & 15;
  v8f acc = {0.f, 0.f, 0.f, 0.f, 0.f, 0.f, 0.f, 0.f};

  for (int kb = wave; kb < KDIM / 32; kb += 8) {
    const int kbase = kb * 32;
    // A fragment (16x32 f16): row M = l15,
    // lanes 0-15 hold K 0..15, lanes 16-31 hold K 16..31
    const int K0 = kbase + half * 16;
    const _Float16* arow = sF + l15 * KDIM;
    v8h alo = *(const v8h*)(arow + K0);        // ds_load_b128
    v8h ahi = *(const v8h*)(arow + K0 + 8);    // ds_load_b128
    v16h af;
    ((v8h*)&af)[0] = alo;
    ((v8h*)&af)[1] = ahi;

    // B fragment (32x16): column N = l15 (W row), lanes 0-15 hold K 0..15,
    // lanes 16-31 hold K 16..31; rows >= 10 are zero padding.
    const int KB0 = kbase + half * 16;
    v16h bf;
    if (l15 < 10) {
      const float* wr = W + (size_t)l15 * KDIM + KB0;
      #pragma unroll
      for (int i = 0; i < 16; ++i) bf[i] = (_Float16)wr[i];
    } else {
      #pragma unroll
      for (int i = 0; i < 16; ++i) bf[i] = (_Float16)0.f;
    }

    acc = __builtin_amdgcn_wmma_f32_16x16x32_f16(
        /*neg_a=*/false, af, /*neg_b=*/false, bf,
        /*c_mod=*/(short)0, acc, /*reuse_a=*/false, /*reuse_b=*/false);
  }

  // C/D layout: VGPR r -> row r (lanes 0-15) / r+8 (lanes 16-31), col = l15
  #pragma unroll
  for (int r = 0; r < 8; ++r)
    sAcc[wave][r + half * 8][l15] = acc[r];
  __syncthreads();

  // ---- Phase 4: reduce the 8 partial tiles, add bias ----
  {
    int row = tid >> 4, col = tid & 15;
    float s = 0.f;
    #pragma unroll
    for (int w = 0; w < 8; ++w) s += sAcc[w][row][col];
    if (col < 10) s += bias[col];
    sLog[row][col] = s;
  }
  __syncthreads();

  // ---- Phase 5: log_softmax over the 10 valid columns ----
  if (tid < NROW) {
    float m = -3.4e38f;
    #pragma unroll
    for (int n = 0; n < 10; ++n) m = fmaxf(m, sLog[tid][n]);
    float se = 0.f;
    #pragma unroll
    for (int n = 0; n < 10; ++n) se += __expf(sLog[tid][n] - m);
    float lse = m + __logf(se);
    float* o = out + (size_t)(b0 + tid) * 10;
    #pragma unroll
    for (int n = 0; n < 10; ++n) o[n] = sLog[tid][n] - lse;
  }
}

extern "C" void kernel_launch(void* const* d_in, const int* in_sizes, int n_in,
                              void* d_out, int out_size, void* d_ws, size_t ws_size,
                              hipStream_t stream) {
  const float* x      = (const float*)d_in[0];   // (B,1,28,28)
  const float* params = (const float*)d_in[1];   // (4,4)
  const float* W      = (const float*)d_in[2];   // (10,1568)
  const float* bias   = (const float*)d_in[3];   // (10,)
  float* out = (float*)d_out;                    // (B,10)

  const int B = in_sizes[0] / 784;               // 2048
  const int grid = B / NROW;                     // 128 workgroups
  quanv_fused_kernel<<<grid, 256, 0, stream>>>(x, params, W, bias, out);
}
